// MemoryRouter_8143257993987
// MI455X (gfx1250) — matrix operand order repaired and checked by reading, not compiled, not run on hardware
//
#include <hip/hip_runtime.h>
#include <cstdint>

// Problem constants (match the reference).
#define B_  4
#define S_  2048
#define D_  2048
#define DC_ 256
#define K_  64
#define M_  2048

typedef __attribute__((ext_vector_type(16))) __bf16 v16bf;
typedef __attribute__((ext_vector_type(2)))  __bf16 v2bf;
typedef __attribute__((ext_vector_type(8)))  float  v8f;
static_assert(sizeof(v16bf) == 32, "v16bf must be 32 bytes");

// Pack two f32 into two bf16 (RNE) in one dword (native gfx1250 cvt).
__device__ __forceinline__ unsigned pack_bf16(float lo, float hi) {
  union { v2bf v; unsigned u; } c;
  c.v = (v2bf){(__bf16)lo, (__bf16)hi};
  return c.u;
}

__device__ __forceinline__ float warpSum(float v) {
#pragma unroll
  for (int o = 16; o > 0; o >>= 1) v += __shfl_xor(v, o, 32);
  return v;
}
__device__ __forceinline__ float warpMax(float v) {
#pragma unroll
  for (int o = 16; o > 0; o >>= 1) v = fmaxf(v, __shfl_xor(v, o, 32));
  return v;
}

// ---------------------------------------------------------------------------
// Generic batched GEMM:  C = act(A @ B + bias), specialized on <TA,TB,ACT>.
//   A: [Mr x Kd] (lda) or, if TA, stored [Kd x Mr] (lda = row stride)
//   B: [Kd x N]  (ldb) or, if TB, stored [N x Kd] row-major (ldb = row stride)
// Tile 64x64, 256 threads = 8 waves; wave = 16 rows x 32 cols
//   => two v_wmma_f32_16x16x32_bf16 per 32-wide k-step.
// Software pipeline: register prefetch of tile i+1 + double-buffered LDS,
// one barrier per k-step; global loads overlap the WMMAs of the prior tile.
// Requires: Mr%64==0, N%64==0, Kd%32==0, lda/ldb multiples of 8, 16B-aligned.
// ---------------------------------------------------------------------------
#define LDS_STRIDE 40   // bf16/row: 80B rows -> 16B-aligned packed stores

template <int TA, int TB, int ACT>
__global__ __launch_bounds__(256)
void mr_gemm(const float* __restrict__ A, const float* __restrict__ Bm,
             const float* __restrict__ bias, float* __restrict__ C,
             int Mr, int N, int Kd,
             long sA, long sB, long sC,
             int lda, int ldb, int ldc)
{
  __shared__ __bf16 lsA[2][64 * LDS_STRIDE];   // A tile [m][k], double-buffered
  __shared__ __bf16 lsB[2][64 * LDS_STRIDE];   // B tile [n][k], double-buffered

  const int tid = threadIdx.x;
  const int bz  = blockIdx.z;
  A  += (size_t)bz * sA;
  Bm += (size_t)bz * sB;
  C  += (size_t)bz * sC;

  const int n0 = blockIdx.x * 64;
  const int m0 = blockIdx.y * 64;
  const int wid = tid >> 5, lane = tid & 31;
  const int l15 = lane & 15;
  const int hi  = (lane >> 4) & 1;
  const int waveM = wid & 3;    // 4 row-groups of 16
  const int waveN = wid >> 2;   // 2 col-groups of 32

  // Per-thread staging coordinates (compile-time selected).
  const int aM  = TA ? 0          : (tid >> 2);
  const int aK  = TA ? (tid >> 3) : 0;
  const int aS  = TA ? (tid & 7) * 8 : (tid & 3) * 8;  // ms or ks
  const int bN  = TB ? (tid >> 2) : 0;
  const int bK  = TB ? 0          : (tid >> 3);
  const int bS  = TB ? (tid & 3) * 8 : (tid & 7) * 8;  // ks or ns

  auto loadA = [&](int k0, float4& r0, float4& r1) {
    const float* ap = TA ? (A + (size_t)(k0 + aK) * lda + m0 + aS)
                         : (A + (size_t)(m0 + aM) * lda + k0 + aS);
    r0 = *(const float4*)ap;
    r1 = *(const float4*)(ap + 4);
  };
  auto loadB = [&](int k0, float4& r0, float4& r1) {
    const float* bp = TB ? (Bm + (size_t)(n0 + bN) * ldb + k0 + bS)
                         : (Bm + (size_t)(bK + k0) * ldb + n0 + bS);
    r0 = *(const float4*)bp;
    r1 = *(const float4*)(bp + 4);
  };
  auto storeA = [&](int cur, float4 r0, float4 r1) {
    if constexpr (!TA) {               // contiguous: one ds_store_b128
      uint4 q;
      q.x = pack_bf16(r0.x, r0.y);
      q.y = pack_bf16(r0.z, r0.w);
      q.z = pack_bf16(r1.x, r1.y);
      q.w = pack_bf16(r1.z, r1.w);
      *(uint4*)&lsA[cur][aM * LDS_STRIDE + aS] = q;
    } else {                           // transposed scatter
      __bf16* d = &lsA[cur][aS * LDS_STRIDE + aK];
      d[0 * LDS_STRIDE] = (__bf16)r0.x;
      d[1 * LDS_STRIDE] = (__bf16)r0.y;
      d[2 * LDS_STRIDE] = (__bf16)r0.z;
      d[3 * LDS_STRIDE] = (__bf16)r0.w;
      d[4 * LDS_STRIDE] = (__bf16)r1.x;
      d[5 * LDS_STRIDE] = (__bf16)r1.y;
      d[6 * LDS_STRIDE] = (__bf16)r1.z;
      d[7 * LDS_STRIDE] = (__bf16)r1.w;
    }
  };
  auto storeB = [&](int cur, float4 r0, float4 r1) {
    if constexpr (TB) {                // contiguous: one ds_store_b128
      uint4 q;
      q.x = pack_bf16(r0.x, r0.y);
      q.y = pack_bf16(r0.z, r0.w);
      q.z = pack_bf16(r1.x, r1.y);
      q.w = pack_bf16(r1.z, r1.w);
      *(uint4*)&lsB[cur][bN * LDS_STRIDE + bS] = q;
    } else {                           // transposed scatter
      __bf16* d = &lsB[cur][bS * LDS_STRIDE + bK];
      d[0 * LDS_STRIDE] = (__bf16)r0.x;
      d[1 * LDS_STRIDE] = (__bf16)r0.y;
      d[2 * LDS_STRIDE] = (__bf16)r0.z;
      d[3 * LDS_STRIDE] = (__bf16)r0.w;
      d[4 * LDS_STRIDE] = (__bf16)r1.x;
      d[5 * LDS_STRIDE] = (__bf16)r1.y;
      d[6 * LDS_STRIDE] = (__bf16)r1.z;
      d[7 * LDS_STRIDE] = (__bf16)r1.w;
    }
  };

  v8f acc0 = {};
  v8f acc1 = {};

  float4 ra0, ra1, rb0, rb1;
  loadA(0, ra0, ra1);
  loadB(0, rb0, rb1);

  int cur = 0;
  for (int k0 = 0; k0 < Kd; k0 += 32) {
    storeA(cur, ra0, ra1);
    storeB(cur, rb0, rb1);
    __syncthreads();                   // single barrier per k-step
    if (k0 + 32 < Kd) {                // prefetch next tile (overlaps WMMAs)
      loadA(k0 + 32, ra0, ra1);
      loadB(k0 + 32, rb0, rb1);
    }

    // ---- build fragments per ISA 16-bit layouts and issue WMMA ----
    union { v16bf v; unsigned u[8]; } Af, Bf0, Bf1;
    const int abase = (waveM * 16 + l15) * LDS_STRIDE;
#pragma unroll
    for (int i = 0; i < 8; ++i) {
      const int kk = (i < 4 ? 2 * i : 16 + 2 * (i - 4)) + (hi ? 8 : 0);
      Af.u[i] = *(const unsigned*)&lsA[cur][abase + kk];
    }
    const int b0base = (waveN * 32 + l15) * LDS_STRIDE;
    const int b1base = b0base + 16 * LDS_STRIDE;
#pragma unroll
    for (int i = 0; i < 8; ++i) {
      const int kk = (i < 4 ? 2 * i : 16 + 2 * (i - 4)) + (hi ? 8 : 0);
      Bf0.u[i] = *(const unsigned*)&lsB[cur][b0base + kk];
      Bf1.u[i] = *(const unsigned*)&lsB[cur][b1base + kk];
    }
    acc0 = __builtin_amdgcn_wmma_f32_16x16x32_bf16(false, Af.v, false, Bf0.v,
                                                   (short)0, acc0, false, false);
    acc1 = __builtin_amdgcn_wmma_f32_16x16x32_bf16(false, Af.v, false, Bf1.v,
                                                   (short)0, acc1, false, false);
    cur ^= 1;
  }

  // ---- epilogue: bias + optional SiLU, C layout per ISA (lane,VGPR)->(N,M) ----
  const int colBase = n0 + waveN * 32 + l15;
  const int rowBase = m0 + waveM * 16 + hi * 8;
#pragma unroll
  for (int j = 0; j < 2; ++j) {
    const int col = colBase + j * 16;
    const float bv = bias ? bias[col] : 0.0f;
    float* cp = C + (size_t)rowBase * ldc + col;
    v8f a = j ? acc1 : acc0;
#pragma unroll
    for (int v = 0; v < 8; ++v) {
      float val = a[v] + bv;
      if constexpr (ACT == 1) val = val / (1.0f + __expf(-val));   // SiLU
      cp[(size_t)v * ldc] = val;
    }
  }
}

// ---------------------------------------------------------------------------
// Row-wise clip + softmax, in place (score matrix stays L2-resident: 64MB<192MB)
// ---------------------------------------------------------------------------
__global__ __launch_bounds__(256)
void mr_softmax_clip(float* __restrict__ s, int cols, float scale, float clipv)
{
  const int row = blockIdx.x;
  float* p = s + (size_t)row * cols;
  const int tid = threadIdx.x, lane = tid & 31, wid = tid >> 5;
  __shared__ float redM[8], redS[8];
  float v[8]; int ne = 0;
  for (int j = tid; j < cols; j += 256) {
    float x = p[j] * scale;
    v[ne++] = fminf(fmaxf(x, -clipv), clipv);
  }
  float mx = -3.4e38f;
  for (int i = 0; i < ne; ++i) mx = fmaxf(mx, v[i]);
  mx = warpMax(mx);
  if (lane == 0) redM[wid] = mx;
  __syncthreads();
  float gm = redM[0];
  for (int i = 1; i < 8; ++i) gm = fmaxf(gm, redM[i]);
  float sm = 0.f;
  for (int i = 0; i < ne; ++i) { v[i] = __expf(v[i] - gm); sm += v[i]; }
  sm = warpSum(sm);
  if (lane == 0) redS[wid] = sm;
  __syncthreads();
  float gs = 0.f;
  for (int i = 0; i < 8; ++i) gs += redS[i];
  const float inv = 1.0f / gs;
  ne = 0;
  for (int j = tid; j < cols; j += 256) p[j] = v[ne++] * inv;
}

// ---------------------------------------------------------------------------
// LayerNorm over concat([A_row, B_row]); optionally also emit the raw concat.
// ---------------------------------------------------------------------------
__global__ __launch_bounds__(256)
void mr_ln_concat(const float* __restrict__ A, int LA,
                  const float* __restrict__ Bp, int LB,
                  float* __restrict__ lnOut, float* __restrict__ rawOut)
{
  const int row = blockIdx.x;
  const int L = LA + LB;
  const int tid = threadIdx.x, lane = tid & 31, wid = tid >> 5;
  __shared__ float r1[8], r2[8];
  float v[9]; int ne = 0;
  for (int j = tid; j < L; j += 256) {
    v[ne++] = (j < LA) ? A[(size_t)row * LA + j]
                       : Bp[(size_t)row * LB + (j - LA)];
  }
  float s = 0.f;
  for (int i = 0; i < ne; ++i) s += v[i];
  s = warpSum(s);
  if (lane == 0) r1[wid] = s;
  __syncthreads();
  float tot = 0.f;
  for (int i = 0; i < 8; ++i) tot += r1[i];
  const float mean = tot / (float)L;
  float q = 0.f;
  for (int i = 0; i < ne; ++i) { float d = v[i] - mean; q += d * d; }
  q = warpSum(q);
  if (lane == 0) r2[wid] = q;
  __syncthreads();
  float tq = 0.f;
  for (int i = 0; i < 8; ++i) tq += r2[i];
  const float inv = rsqrtf(tq / (float)L + 1e-5f);
  ne = 0;
  for (int j = tid; j < L; j += 256) {
    const float x = v[ne++];
    lnOut[(size_t)row * L + j] = (x - mean) * inv;
    if (rawOut) rawOut[(size_t)row * L + j] = x;
  }
}

// ---------------------------------------------------------------------------
// Per-token gemv + gate. mode 0: straight-through hard gate (fwd = logit>0),
// mode 1: sigmoid.
// ---------------------------------------------------------------------------
__global__ __launch_bounds__(256)
void mr_gemv_gate(const float* __restrict__ H, int Kd,
                  const float* __restrict__ w, const float* __restrict__ b,
                  float* __restrict__ out, int mode)
{
  const int row = blockIdx.x;
  const int tid = threadIdx.x, lane = tid & 31, wid = tid >> 5;
  __shared__ float r1[8];
  const float* h = H + (size_t)row * Kd;
  float s = 0.f;
  for (int j = tid; j < Kd; j += 256) s += h[j] * w[j];
  s = warpSum(s);
  if (lane == 0) r1[wid] = s;
  __syncthreads();
  if (tid == 0) {
    float tot = 0.f;
    for (int i = 0; i < 8; ++i) tot += r1[i];
    const float logit = tot + b[0];
    out[row] = (mode == 0) ? (logit > 0.0f ? 1.0f : 0.0f)
                           : 1.0f / (1.0f + __expf(-logit));
  }
}

// x_fused = x + read_gate * context
__global__ __launch_bounds__(256)
void mr_fuse(const float* __restrict__ x, const float* __restrict__ rg,
             const float* __restrict__ ctxt, float* __restrict__ xf)
{
  const int t = blockIdx.x;
  const float g = rg[t];
  const size_t base = (size_t)t * D_;
  for (int j = threadIdx.x; j < D_; j += 256)
    xf[base + j] = x[base + j] + g * ctxt[base + j];
}

// weighted = t_cache * iw ; write_scores = imp*wg ; iw = max(imp*wg, 1e-6)
__global__ __launch_bounds__(256)
void mr_weighted(const float* __restrict__ imp, const float* __restrict__ wg,
                 const float* __restrict__ tc, float* __restrict__ wtd,
                 float* __restrict__ wsOut, float* __restrict__ iwOut)
{
  const int t = blockIdx.x;
  const float msc = imp[t] * wg[t];
  const float iw = fmaxf(msc, 1e-6f);
  const int c = threadIdx.x;           // blockDim.x == DC_
  wtd[(size_t)t * DC_ + c] = tc[(size_t)t * DC_ + c] * iw;
  if (c == 0) { wsOut[t] = msc; iwOut[t] = iw; }
}

// Gumbel-softmax over K_=64 slots; one wave per token (8 tokens/block).
__global__ __launch_bounds__(256)
void mr_gumbel(const float* __restrict__ slog, const float* __restrict__ u,
               const float* __restrict__ wg, float* __restrict__ msp)
{
  const int tid = threadIdx.x, lane = tid & 31, wid = tid >> 5;
  const int row = blockIdx.x * 8 + wid;
  const size_t base = (size_t)row * K_;
  const float scale = 0.0625f;         // 1/sqrt(DC)
  float l0 = slog[base + lane]      * scale;
  float l1 = slog[base + lane + 32] * scale;
  l0 = fminf(fmaxf(l0, -10.f), 10.f);
  l1 = fminf(fmaxf(l1, -10.f), 10.f);
  const float v0 = l0 - __logf(-__logf(u[base + lane]));
  const float v1 = l1 - __logf(-__logf(u[base + lane + 32]));
  const float mx = warpMax(fmaxf(v0, v1));
  const float e0 = __expf(v0 - mx), e1 = __expf(v1 - mx);
  const float sm = warpSum(e0 + e1);
  const float g = wg[row] / sm;        // soft probs * write_gate
  msp[base + lane]      = e0 * g;
  msp[base + lane + 32] = e1 * g;
}

// slot_w[b,k] = sum_t msp[b,t,k] * iw[b,t]
__global__ __launch_bounds__(256)
void mr_slotw(const float* __restrict__ msp, const float* __restrict__ iw,
              float* __restrict__ sw)
{
  const int bk = blockIdx.x;
  const int b = bk / K_, k = bk % K_;
  const int tid = threadIdx.x, lane = tid & 31, wid = tid >> 5;
  __shared__ float r1[8];
  float s = 0.f;
  for (int t = tid; t < S_; t += 256) {
    const int row = b * S_ + t;
    s += msp[(size_t)row * K_ + k] * iw[row];
  }
  s = warpSum(s);
  if (lane == 0) r1[wid] = s;
  __syncthreads();
  if (tid == 0) {
    float tot = 0.f;
    for (int i = 0; i < 8; ++i) tot += r1[i];
    sw[bk] = tot;
  }
}

// new_local_cache = has ? slot_writes/clip(slot_w,0.1) : local_cache
__global__ __launch_bounds__(256)
void mr_finalize(const float* __restrict__ swr, const float* __restrict__ sw,
                 const float* __restrict__ lc, float* __restrict__ out)
{
  const int bk = blockIdx.x;
  const float w = sw[bk];
  const float inv = 1.0f / fmaxf(w, 0.1f);
  const size_t idx = (size_t)bk * DC_ + threadIdx.x;
  out[idx] = (w > 0.1f) ? swr[idx] * inv : lc[idx];
}

// ---------------------------------------------------------------------------
extern "C" void kernel_launch(void* const* d_in, const int* in_sizes, int n_in,
                              void* d_out, int out_size, void* d_ws, size_t ws_size,
                              hipStream_t stream)
{
  const float* x      = (const float*)d_in[0];
  const float* cache  = (const float*)d_in[1];
  const float* lcache = (const float*)d_in[2];
  const float* gu     = (const float*)d_in[3];
  const float* W_tc = (const float*)d_in[4];  const float* b_tc = (const float*)d_in[5];
  const float* W_fc = (const float*)d_in[6];  const float* b_fc = (const float*)d_in[7];
  const float* W_r1 = (const float*)d_in[8];  const float* b_r1 = (const float*)d_in[9];
  const float* W_r2 = (const float*)d_in[10]; const float* b_r2 = (const float*)d_in[11];
  const float* W_w1 = (const float*)d_in[12]; const float* b_w1 = (const float*)d_in[13];
  const float* W_w2 = (const float*)d_in[14]; const float* b_w2 = (const float*)d_in[15];
  const float* W_i1 = (const float*)d_in[16]; const float* b_i1 = (const float*)d_in[17];
  const float* W_i2 = (const float*)d_in[18]; const float* b_i2 = (const float*)d_in[19];

  float* out_xf = (float*)d_out;                            // [B,S,D]
  float* out_rg = out_xf + (size_t)B_ * S_ * D_;            // [B,S,1]
  float* out_lc = out_rg + (size_t)B_ * S_;                 // [B,K,DC]
  float* out_ws = out_lc + (size_t)B_ * K_ * DC_;           // [B,S]

  // Workspace carve-out (~313 MB of f32; every buffer is written before read).
  float* w = (float*)d_ws;
  size_t off = 0;
  auto take = [&](size_t n) {
    float* p = w + off;
    off += (n + 63) & ~(size_t)63;
    return p;
  };
  float* bufS   = take((size_t)B_ * S_ * M_);          // scores/probs (both phases)
  float* bufXC  = take((size_t)B_ * S_ * DC_);         // x_cache, then t_cache
  float* bufCTX = take((size_t)B_ * S_ * DC_);         // ctx_c, then ctx2
  float* bufLN  = take((size_t)B_ * S_ * (D_ + DC_));  // ln(comb) / ln(comb2)
  float* bufH   = take((size_t)B_ * S_ * (D_ / 2));    // h, then hw
  float* bufHI  = take((size_t)B_ * S_ * (D_ / 2));    // hi
  float* bufBIG = take((size_t)B_ * S_ * (D_ + DC_));  // context, then comb2 raw
  float* bufWG  = take((size_t)B_ * S_);
  float* bufIMP = take((size_t)B_ * S_);
  float* bufIW  = take((size_t)B_ * S_);
  float* bufSL  = take((size_t)B_ * S_ * K_);          // slot logits
  float* bufMSP = take((size_t)B_ * S_ * K_);          // masked slot probs
  float* bufWTD = take((size_t)B_ * S_ * DC_);         // weighted t_cache
  float* bufSWR = take((size_t)B_ * K_ * DC_);         // slot writes (raw)
  float* bufSW  = take(B_ * K_);                       // slot weights

  const dim3 blk(256);
  const int BS = B_ * S_;

  // Template dispatch over the 4 used (TA,TB,ACT) combos.
  auto gemm = [&](int tA, int tB, int act,
                  const float* A, const float* Bm, const float* bias, float* C,
                  int Mr, int N, int Kd, long sA, long sB, long sC,
                  int lda, int ldb, int ldc, int nb) {
    dim3 g(N / 64, Mr / 64, nb);
    if (tA == 0 && tB == 0 && act == 0)
      mr_gemm<0,0,0><<<g, blk, 0, stream>>>(A, Bm, bias, C, Mr, N, Kd, sA, sB, sC, lda, ldb, ldc);
    else if (tA == 0 && tB == 0 && act == 1)
      mr_gemm<0,0,1><<<g, blk, 0, stream>>>(A, Bm, bias, C, Mr, N, Kd, sA, sB, sC, lda, ldb, ldc);
    else if (tA == 0 && tB == 1)
      mr_gemm<0,1,0><<<g, blk, 0, stream>>>(A, Bm, bias, C, Mr, N, Kd, sA, sB, sC, lda, ldb, ldc);
    else
      mr_gemm<1,0,0><<<g, blk, 0, stream>>>(A, Bm, bias, C, Mr, N, Kd, sA, sB, sC, lda, ldb, ldc);
  };

  // ================= READ phase =================
  gemm(0,0,0, x, W_tc, b_tc, bufXC, BS, DC_, D_, 0, 0, 0, D_, DC_, DC_, 1);
  gemm(0,1,0, bufXC, cache, nullptr, bufS, S_, M_, DC_,
       (long)S_ * DC_, (long)M_ * DC_, (long)S_ * M_, DC_, DC_, M_, B_);
  mr_softmax_clip<<<dim3(BS), blk, 0, stream>>>(bufS, M_, 0.0625f, 20.0f);
  gemm(0,0,0, bufS, cache, nullptr, bufCTX, S_, DC_, M_,
       (long)S_ * M_, (long)M_ * DC_, (long)S_ * DC_, M_, DC_, DC_, B_);
  mr_ln_concat<<<dim3(BS), blk, 0, stream>>>(bufXC, DC_, bufCTX, DC_, bufLN, nullptr);
  gemm(0,0,1, bufLN, W_r1, b_r1, bufH, BS, D_ / 2, 2 * DC_,
       0, 0, 0, 2 * DC_, D_ / 2, D_ / 2, 1);
  mr_gemv_gate<<<dim3(BS), blk, 0, stream>>>(bufH, D_ / 2, W_r2, b_r2, out_rg, 0);
  gemm(0,0,0, bufCTX, W_fc, b_fc, bufBIG, BS, D_, DC_, 0, 0, 0, DC_, D_, D_, 1);
  mr_fuse<<<dim3(BS), blk, 0, stream>>>(x, out_rg, bufBIG, out_xf);

  // ================= WRITE phase =================
  gemm(0,0,0, out_xf, W_tc, b_tc, bufXC, BS, DC_, D_, 0, 0, 0, D_, DC_, DC_, 1);
  gemm(0,1,0, bufXC, cache, nullptr, bufS, S_, M_, DC_,
       (long)S_ * DC_, (long)M_ * DC_, (long)S_ * M_, DC_, DC_, M_, B_);
  mr_softmax_clip<<<dim3(BS), blk, 0, stream>>>(bufS, M_, 0.0625f, 20.0f);
  gemm(0,0,0, bufS, cache, nullptr, bufCTX, S_, DC_, M_,
       (long)S_ * M_, (long)M_ * DC_, (long)S_ * DC_, M_, DC_, DC_, B_);
  mr_ln_concat<<<dim3(BS), blk, 0, stream>>>(out_xf, D_, bufCTX, DC_, bufLN, bufBIG);
  gemm(0,0,1, bufLN, W_w1, b_w1, bufH, BS, D_ / 2, D_ + DC_,
       0, 0, 0, D_ + DC_, D_ / 2, D_ / 2, 1);
  mr_gemv_gate<<<dim3(BS), blk, 0, stream>>>(bufH, D_ / 2, W_w2, b_w2, bufWG, 0);
  gemm(0,0,1, bufBIG, W_i1, b_i1, bufHI, BS, D_ / 2, D_ + DC_,
       0, 0, 0, D_ + DC_, D_ / 2, D_ / 2, 1);
  mr_gemv_gate<<<dim3(BS), blk, 0, stream>>>(bufHI, D_ / 2, W_i2, b_i2, bufIMP, 1);
  mr_weighted<<<dim3(BS), blk, 0, stream>>>(bufIMP, bufWG, bufXC, bufWTD, out_ws, bufIW);
  gemm(0,1,0, bufXC, lcache, nullptr, bufSL, S_, K_, DC_,
       (long)S_ * DC_, (long)K_ * DC_, (long)S_ * K_, DC_, DC_, K_, B_);
  mr_gumbel<<<dim3(BS / 8), blk, 0, stream>>>(bufSL, gu, bufWG, bufMSP);
  gemm(1,0,0, bufMSP, bufWTD, nullptr, bufSWR, K_, DC_, S_,
       (long)S_ * K_, (long)S_ * DC_, (long)K_ * DC_, K_, DC_, DC_, B_);
  mr_slotw<<<dim3(B_ * K_), blk, 0, stream>>>(bufMSP, bufIW, bufSW);
  mr_finalize<<<dim3(B_ * K_), blk, 0, stream>>>(bufSWR, bufSW, lcache, out_lc);
}
